// ContrastiveClusterLoss_76459007803418
// MI455X (gfx1250) — compile-verified, matrix operand order checked by reading, not run on hardware
//
#include <hip/hip_runtime.h>

// ContrastiveClusterLoss for MI455X (gfx1250, wave32).
// GEMM via V_WMMA_F32_16X16X32_BF16 with bf16x3 split (hi*hi + hi*lo + lo*hi)
// for ~f32 accuracy on the fast XDL path. Norms kept in exact f32.
// Fused margin-loss epilogue, f64 atomic accumulation.
// ws layout: [0..15] double accum, then a2[B], b2[K] (floats).

typedef __attribute__((ext_vector_type(16))) __bf16 v16bf;
typedef __attribute__((ext_vector_type(8)))  __bf16 v8bf;
typedef __attribute__((ext_vector_type(4)))  __bf16 v4bf;
typedef __attribute__((ext_vector_type(8)))  float  v8f;

#define MARGIN 1.0f
#define BM 64     // macro-tile rows (embeddings)
#define BN 128    // macro-tile cols (centroids)
#define DK 32     // K-chunk along D == bf16 WMMA K depth
#define BSTR 40   // padded bf16 row stride: 80B rows -> 16B-aligned b128 reads,
                  // 20-dword stride -> each half-wave b128 read covers all 64 banks

__global__ void ccl_init(double* accum) { *accum = 0.0; }

// one wave per row: ||x_row||^2 in exact f32
__global__ void ccl_row_norms(const float* __restrict__ X, float* __restrict__ out,
                              int rows, int D) {
    int row  = blockIdx.x * (blockDim.x >> 5) + (threadIdx.x >> 5);
    int lane = threadIdx.x & 31;
    if (row >= rows) return;
    const float* p = X + (size_t)row * D;
    float s = 0.f;
    for (int i = lane * 4; i < D; i += 128) {
        float4 v = *(const float4*)(p + i);
        s += v.x * v.x + v.y * v.y + v.z * v.z + v.w * v.w;
    }
#pragma unroll
    for (int off = 16; off > 0; off >>= 1) s += __shfl_down(s, off);
    if (lane == 0) out[row] = s;
}

// Build a v16bf A/B fragment for 16x16x32 bf16 WMMA from an LDS row.
// ISA layout: lane L -> VGPR0..3 = K{8h..8h+7}, VGPR4..7 = K{16+8h..16+8h+7}, h=L/16.
__device__ __forceinline__ v16bf ccl_frag(const __bf16* rowp, int h) {
    v8bf klo = *(const v8bf*)(rowp + 8 * h);        // K = 8h .. 8h+7
    v8bf khi = *(const v8bf*)(rowp + 16 + 8 * h);   // K = 16+8h .. 16+8h+7
    v16bf r;
#pragma unroll
    for (int i = 0; i < 8; ++i) { r[i] = klo[i]; r[8 + i] = khi[i]; }
    return r;
}

__global__ __launch_bounds__(256) void ccl_gemm_loss(
    const float* __restrict__ E,      // [B,D] embeddings
    const float* __restrict__ Cc,     // [K,D] centroids
    const int*   __restrict__ labels, // [B]
    const float* __restrict__ a2,     // [B] row norms^2
    const float* __restrict__ b2,     // [K] centroid norms^2
    double* __restrict__ accum,
    int B, int K, int D) {
    __shared__ __bf16 Ah[BM][BSTR], Al[BM][BSTR];
    __shared__ __bf16 Bh[BN][BSTR], Bl[BN][BSTR];

    const int tid  = threadIdx.x;
    const int lane = tid & 31;
    const int wave = tid >> 5;
    const int bm   = blockIdx.x * BM;
    const int bn   = blockIdx.y * BN;
    const int mW   = (wave & 1) * 32;   // 2 waves along M
    const int nW   = (wave >> 1) * 32;  // 4 waves along N

    v8f acc00 = {}, acc01 = {}, acc10 = {}, acc11 = {};

    for (int d0 = 0; d0 < D; d0 += DK) {
        __syncthreads();
        // A tile: 64x32 f32 = 512 float4; 256 threads -> 2 each; split hi/lo bf16
#pragma unroll
        for (int it = 0; it < (BM * DK / 4) / 256; ++it) {
            int i = tid + it * 256;
            int row = i >> 3, c4 = i & 7;  // 8 float4 per row
            float4 v = *(const float4*)(E + (size_t)(bm + row) * D + d0 + c4 * 4);
            v4bf h = { (__bf16)v.x, (__bf16)v.y, (__bf16)v.z, (__bf16)v.w };
            v4bf l = { (__bf16)(v.x - (float)h[0]), (__bf16)(v.y - (float)h[1]),
                       (__bf16)(v.z - (float)h[2]), (__bf16)(v.w - (float)h[3]) };
            *(v4bf*)&Ah[row][c4 * 4] = h;
            *(v4bf*)&Al[row][c4 * 4] = l;
        }
        // B tile: 128x32 f32 = 1024 float4; 4 each
#pragma unroll
        for (int it = 0; it < (BN * DK / 4) / 256; ++it) {
            int i = tid + it * 256;
            int row = i >> 3, c4 = i & 7;
            float4 v = *(const float4*)(Cc + (size_t)(bn + row) * D + d0 + c4 * 4);
            v4bf h = { (__bf16)v.x, (__bf16)v.y, (__bf16)v.z, (__bf16)v.w };
            v4bf l = { (__bf16)(v.x - (float)h[0]), (__bf16)(v.y - (float)h[1]),
                       (__bf16)(v.z - (float)h[2]), (__bf16)(v.w - (float)h[3]) };
            *(v4bf*)&Bh[row][c4 * 4] = h;
            *(v4bf*)&Bl[row][c4 * 4] = l;
        }
        __syncthreads();

        const int h  = lane >> 4;
        const int mr = mW + (lane & 15);
        const int nc = nW + (lane & 15);
        v16bf a0h = ccl_frag(&Ah[mr][0], h),      a0l = ccl_frag(&Al[mr][0], h);
        v16bf a1h = ccl_frag(&Ah[mr + 16][0], h), a1l = ccl_frag(&Al[mr + 16][0], h);
        v16bf b0h = ccl_frag(&Bh[nc][0], h),      b0l = ccl_frag(&Bl[nc][0], h);
        v16bf b1h = ccl_frag(&Bh[nc + 16][0], h), b1l = ccl_frag(&Bl[nc + 16][0], h);

#define CCL_WMMA(A_, B_, C_) \
        C_ = __builtin_amdgcn_wmma_f32_16x16x32_bf16(false, A_, false, B_, (short)0, C_, false, false)
        // bf16x3: hi*hi + hi*lo + lo*hi (lo*lo ~ 2^-18 rel, dropped)
        CCL_WMMA(a0h, b0h, acc00); CCL_WMMA(a0h, b1h, acc01);
        CCL_WMMA(a1h, b0h, acc10); CCL_WMMA(a1h, b1h, acc11);
        CCL_WMMA(a0h, b0l, acc00); CCL_WMMA(a0h, b1l, acc01);
        CCL_WMMA(a1h, b0l, acc10); CCL_WMMA(a1h, b1l, acc11);
        CCL_WMMA(a0l, b0h, acc00); CCL_WMMA(a0l, b1h, acc01);
        CCL_WMMA(a1l, b0h, acc10); CCL_WMMA(a1l, b1h, acc11);
#undef CCL_WMMA
    }

    // Fused loss epilogue. C/D layout: VGPR v -> M = v + 8*(lane/16), N = lane%16.
    float lsum = 0.f;
    const int half = lane >> 4;
    const int l15  = lane & 15;
    v8f accs[4] = {acc00, acc01, acc10, acc11};
#pragma unroll
    for (int t = 0; t < 4; ++t) {
        int tm = t >> 1, tn = t & 1;
        int n = bn + nW + tn * 16 + l15;
        float bb = b2[n];
        v8f c = accs[t];
#pragma unroll
        for (int v = 0; v < 8; ++v) {
            int m = bm + mW + tm * 16 + v + half * 8;
            float d2 = a2[m] + bb - 2.0f * c[v];
            d2 = fmaxf(d2, 0.f);
            if (labels[m] == n) {
                lsum += d2;                       // pos: ||e - c_true||^2
            } else {
                float dd = sqrtf(d2);             // neg: relu(margin - d)^2
                float tt = fmaxf(MARGIN - dd, 0.f);
                lsum += tt * tt;
            }
        }
    }
#pragma unroll
    for (int off = 16; off > 0; off >>= 1) lsum += __shfl_down(lsum, off);
    if (lane == 0) atomicAdd(accum, (double)lsum);  // global_atomic_add_f64
}

__global__ void ccl_finalize(const double* accum, float* out, int B) {
    out[0] = (float)(*accum / (double)B);
}

extern "C" void kernel_launch(void* const* d_in, const int* in_sizes, int n_in,
                              void* d_out, int out_size, void* d_ws, size_t ws_size,
                              hipStream_t stream) {
    const float* E      = (const float*)d_in[0];  // [B,D]
    const float* Cc     = (const float*)d_in[1];  // [K,D]
    const int*   labels = (const int*)d_in[2];    // [B]

    const int B = in_sizes[2];
    const int D = in_sizes[0] / B;
    const int K = in_sizes[1] / D;

    double* accum = (double*)d_ws;
    float*  a2    = (float*)((char*)d_ws + 16);
    float*  b2    = a2 + B;

    ccl_init<<<1, 1, 0, stream>>>(accum);
    ccl_row_norms<<<(B + 7) / 8, 256, 0, stream>>>(E, a2, B, D);
    ccl_row_norms<<<(K + 7) / 8, 256, 0, stream>>>(Cc, b2, K, D);

    dim3 grid(B / BM, K / BN);  // 512 x 8 = 4096 workgroups
    ccl_gemm_loss<<<grid, 256, 0, stream>>>(E, Cc, labels, a2, b2, accum, B, K, D);

    ccl_finalize<<<1, 1, 0, stream>>>(accum, (float*)d_out, B);
}
